// HypergraphGuidedChebNet_69724499083636
// MI455X (gfx1250) — matrix operand order repaired and checked by reading, not compiled, not run on hardware
//
#include <hip/hip_runtime.h>

#define N_NODES 50000
#define N_EDGES 10000
#define NNZ     800000
#define F_IN    128
#define HID     512
#define K_CHEB  10

typedef unsigned short u16;
typedef __attribute__((ext_vector_type(16))) __bf16 v16bf;
typedef __attribute__((ext_vector_type(8)))  float  v8f;
typedef __attribute__((ext_vector_type(8)))  u16    v8u;

union V16 { v8u h[2]; v16bf bf; };

__device__ __forceinline__ u16 f2bf(float x) {
  unsigned u = __float_as_uint(x);
  u += 0x7FFFu + ((u >> 16) & 1u);   // round-to-nearest-even
  return (u16)(u >> 16);
}
__device__ __forceinline__ float bf2f(u16 h) {
  return __uint_as_float(((unsigned)h) << 16);
}

// ---- degree accumulation: d_e += 1, d_v += w[e] per (v,e) pair -------------
__global__ __launch_bounds__(256) void k_degrees(const int* __restrict__ nidx,
                                                 const int* __restrict__ eidx,
                                                 const float* __restrict__ ew,
                                                 float* __restrict__ d_e,
                                                 float* __restrict__ d_v) {
  int i = blockIdx.x * 256 + threadIdx.x;            // grid covers NNZ exactly
  int e = eidx[i];
  int v = nidx[i];
  atomicAdd(&d_e[e], 1.0f);
  atomicAdd(&d_v[v], ew[e]);
}

// ---- dv_is = rsqrt(max(d_v,eps)) (0 if d_v<=0); de_w = w/d_e ---------------
__global__ __launch_bounds__(256) void k_norms(const float* __restrict__ d_v,
                                               const float* __restrict__ d_e,
                                               const float* __restrict__ ew,
                                               float* __restrict__ dv_is,
                                               float* __restrict__ de_w) {
  int i = blockIdx.x * 256 + threadIdx.x;
  if (i < N_NODES) {
    float dv = d_v[i];
    dv_is[i] = dv > 0.0f ? rsqrtf(fmaxf(dv, 1e-12f)) : 0.0f;
  }
  if (i < N_EDGES) {
    float de = d_e[i];
    de_w[i] = ew[i] * (de > 0.0f ? 1.0f / fmaxf(de, 1e-12f) : 0.0f);
  }
}

// ---- acc = theta[0] * x (float4 granularity) -------------------------------
__global__ __launch_bounds__(256) void k_init_acc(const float4* __restrict__ x4,
                                                  const float* __restrict__ theta,
                                                  float4* __restrict__ acc4) {
  int i = blockIdx.x * 256 + threadIdx.x;            // grid covers N*F/4 exactly
  float t = theta[0];
  float4 xv = x4[i];
  float4 a; a.x = t * xv.x; a.y = t * xv.y; a.z = t * xv.z; a.w = t * xv.w;
  acc4[i] = a;
}

// ---- scatter node->edge: m_e[e] += dv_is[v] * T[v]  (wave per pair) --------
__global__ __launch_bounds__(256) void k_spmm_v2e(const float* __restrict__ Tin,
                                                  const int* __restrict__ nidx,
                                                  const int* __restrict__ eidx,
                                                  const float* __restrict__ dv_is,
                                                  float* __restrict__ m_e) {
  unsigned gid  = blockIdx.x * 256u + threadIdx.x;   // grid covers NNZ*32 exactly
  unsigned pair = gid >> 5;
  unsigned lane = gid & 31u;
  int v = nidx[pair];
  int e = eidx[pair];
  float s = dv_is[v];
  float4 xv = *reinterpret_cast<const float4*>(Tin + (size_t)v * F_IN + lane * 4);
  float* dst = m_e + (size_t)e * F_IN + lane * 4;
  atomicAdd(dst + 0, s * xv.x);
  atomicAdd(dst + 1, s * xv.y);
  atomicAdd(dst + 2, s * xv.z);
  atomicAdd(dst + 3, s * xv.w);
}

// ---- scatter edge->node: p[v] += de_w[e] * m_e[e]  (wave per pair) ---------
__global__ __launch_bounds__(256) void k_spmm_e2v(const float* __restrict__ m_e,
                                                  const int* __restrict__ nidx,
                                                  const int* __restrict__ eidx,
                                                  const float* __restrict__ de_w,
                                                  float* __restrict__ p) {
  unsigned gid  = blockIdx.x * 256u + threadIdx.x;
  unsigned pair = gid >> 5;
  unsigned lane = gid & 31u;
  int v = nidx[pair];
  int e = eidx[pair];
  float s = de_w[e];
  float4 mv = *reinterpret_cast<const float4*>(m_e + (size_t)e * F_IN + lane * 4);
  float* dst = p + (size_t)v * F_IN + lane * 4;
  atomicAdd(dst + 0, s * mv.x);
  atomicAdd(dst + 1, s * mv.y);
  atomicAdd(dst + 2, s * mv.z);
  atomicAdd(dst + 3, s * mv.w);
}

// ---- k=1: T1 = -dv_is*p ; acc += theta[1]*T1 -------------------------------
__global__ __launch_bounds__(256) void k_combine1(const float4* __restrict__ p4,
                                                  const float* __restrict__ dv_is,
                                                  const float* __restrict__ theta,
                                                  float4* __restrict__ acc4,
                                                  float4* __restrict__ tcur4) {
  int i = blockIdx.x * 256 + threadIdx.x;            // grid covers N*F/4 exactly
  int v = i >> 5;                                    // 32 float4 per row
  float s  = dv_is[v];
  float th = theta[1];
  float4 pv = p4[i];
  float4 t; t.x = -s * pv.x; t.y = -s * pv.y; t.z = -s * pv.z; t.w = -s * pv.w;
  tcur4[i] = t;
  float4 a = acc4[i];
  a.x += th * t.x; a.y += th * t.y; a.z += th * t.z; a.w += th * t.w;
  acc4[i] = a;
}

// ---- k>=2: Tk = -2*dv_is*p - T_{k-2} ; acc += theta[k]*Tk ------------------
__global__ __launch_bounds__(256) void k_combine(const float4* __restrict__ p4,
                                                 const float* __restrict__ dv_is,
                                                 const float* __restrict__ theta, int k,
                                                 const float4* __restrict__ tprev4,
                                                 float4* __restrict__ tnext4,
                                                 float4* __restrict__ acc4) {
  int i = blockIdx.x * 256 + threadIdx.x;
  int v = i >> 5;
  float s  = -2.0f * dv_is[v];
  float th = theta[k];
  float4 pv = p4[i];
  float4 tp = tprev4[i];
  float4 t;
  t.x = s * pv.x - tp.x; t.y = s * pv.y - tp.y;
  t.z = s * pv.z - tp.z; t.w = s * pv.w - tp.w;
  tnext4[i] = t;
  float4 a = acc4[i];
  a.x += th * t.x; a.y += th * t.y; a.z += th * t.z; a.w += th * t.w;
  acc4[i] = a;
}

// ---- f32 -> (bf16 hi, bf16 lo) split ---------------------------------------
__global__ __launch_bounds__(256) void k_bfsplit(const float* __restrict__ in,
                                                 u16* __restrict__ hi,
                                                 u16* __restrict__ lo, int n) {
  int i = blockIdx.x * 256 + threadIdx.x;
  if (i >= n) return;
  float a = in[i];
  u16 h = f2bf(a);
  hi[i] = h;
  lo[i] = f2bf(a - bf2f(h));
}

// ---- WMMA GEMM: out[n,h] = relu(sum_f A[n,f]*W[h,f] + b[h]), bf16x3 --------
// one wave per 16x16 tile; M tiles = 3125, N tiles = 32 (both exact)
__global__ __launch_bounds__(256) void k_gemm_wmma(const u16* __restrict__ Ahi,
                                                   const u16* __restrict__ Alo,
                                                   const u16* __restrict__ Whi,
                                                   const u16* __restrict__ Wlo,
                                                   const float* __restrict__ bias,
                                                   float* __restrict__ out) {
  int wave  = threadIdx.x >> 5;
  int lane  = threadIdx.x & 31;
  int tile  = blockIdx.x * 8 + wave;                 // 12500 blocks * 8 = 100000 tiles
  int mtile = tile >> 5;                             // [0,3125)
  int ntile = tile & 31;                             // [0,32)
  int half  = lane >> 4;
  int l16   = lane & 15;

  const u16* ah = Ahi + (size_t)(mtile * 16 + l16) * F_IN;
  const u16* al = Alo + (size_t)(mtile * 16 + l16) * F_IN;
  const u16* wh = Whi + (size_t)(ntile * 16 + l16) * F_IN;
  const u16* wl = Wlo + (size_t)(ntile * 16 + l16) * F_IN;

  v8f c = {0.f, 0.f, 0.f, 0.f, 0.f, 0.f, 0.f, 0.f};
#pragma unroll
  for (int s = 0; s < 4; ++s) {
    // A 16x32 layout: 16 contiguous K at base 16*half
    int ka0 = 32 * s + 16 * half;
    int ka1 = 32 * s + 16 * half + 8;
    // B 32x16 layout: 16 contiguous K at base 16*half
    int kb  = 32 * s + 16 * half;
    V16 A0, A1, B0, B1;
    A0.h[0] = *reinterpret_cast<const v8u*>(ah + ka0);
    A0.h[1] = *reinterpret_cast<const v8u*>(ah + ka1);
    A1.h[0] = *reinterpret_cast<const v8u*>(al + ka0);
    A1.h[1] = *reinterpret_cast<const v8u*>(al + ka1);
    B0.h[0] = *reinterpret_cast<const v8u*>(wh + kb);
    B0.h[1] = *reinterpret_cast<const v8u*>(wh + kb + 8);
    B1.h[0] = *reinterpret_cast<const v8u*>(wl + kb);
    B1.h[1] = *reinterpret_cast<const v8u*>(wl + kb + 8);
    c = __builtin_amdgcn_wmma_f32_16x16x32_bf16(false, A0.bf, false, B0.bf, (short)0, c, false, false);
    c = __builtin_amdgcn_wmma_f32_16x16x32_bf16(false, A0.bf, false, B1.bf, (short)0, c, false, false);
    c = __builtin_amdgcn_wmma_f32_16x16x32_bf16(false, A1.bf, false, B0.bf, (short)0, c, false, false);
  }

  // C/D layout: lane L, vgpr r -> (M = r + 8*half, N = L&15)
  int n     = ntile * 16 + l16;
  int mbase = mtile * 16 + 8 * half;
  float bv  = bias[n];
  float* op = out + (size_t)mbase * HID + n;
#pragma unroll
  for (int r = 0; r < 8; ++r) {
    float v = c[r] + bv;
    op[(size_t)r * HID] = v > 0.0f ? v : 0.0f;
  }
}

extern "C" void kernel_launch(void* const* d_in, const int* in_sizes, int n_in,
                              void* d_out, int out_size, void* d_ws, size_t ws_size,
                              hipStream_t stream) {
  const float* x     = (const float*)d_in[0];
  const int*   nidx  = (const int*)d_in[1];          // hyperedge_index row 0
  const int*   eidx  = ((const int*)d_in[1]) + NNZ;  // hyperedge_index row 1
  const float* ew    = (const float*)d_in[2];
  const float* theta = (const float*)d_in[3];
  const float* w     = (const float*)d_in[4];
  const float* bias  = (const float*)d_in[5];
  float*       out   = (float*)d_out;

  // ---- carve workspace -----------------------------------------------------
  char* ws = (char*)d_ws;
  size_t off = 0;
  auto carve = [&](size_t bytes) -> void* {
    void* p = ws + off;
    off = (off + bytes + 255) & ~(size_t)255;
    return p;
  };
  float* d_e   = (float*)carve((size_t)N_EDGES * 4);
  float* d_v   = (float*)carve((size_t)N_NODES * 4);
  float* dv_is = (float*)carve((size_t)N_NODES * 4);
  float* de_w  = (float*)carve((size_t)N_EDGES * 4);
  float* m_e   = (float*)carve((size_t)N_EDGES * F_IN * 4);
  float* p     = (float*)carve((size_t)N_NODES * F_IN * 4);
  float* t0    = (float*)carve((size_t)N_NODES * F_IN * 4);
  float* t1    = (float*)carve((size_t)N_NODES * F_IN * 4);
  float* acc   = (float*)carve((size_t)N_NODES * F_IN * 4);
  u16*   a_hi  = (u16*)carve((size_t)N_NODES * F_IN * 2);
  u16*   a_lo  = (u16*)carve((size_t)N_NODES * F_IN * 2);
  u16*   w_hi  = (u16*)carve((size_t)HID * F_IN * 2);
  u16*   w_lo  = (u16*)carve((size_t)HID * F_IN * 2);

  const int NF4      = N_NODES * F_IN / 4;           // 1,600,000
  const int EW_BLKS  = NNZ / 256;                    // 3,125 (exact)
  const int NF4_BLKS = NF4 / 256;                    // 6,250 (exact)
  const int SP_BLKS  = (NNZ * 32) / 256;             // 100,000 (exact)

  // ---- degrees & normalization --------------------------------------------
  hipMemsetAsync(d_e, 0, (size_t)N_EDGES * 4, stream);
  hipMemsetAsync(d_v, 0, (size_t)N_NODES * 4, stream);
  k_degrees<<<EW_BLKS, 256, 0, stream>>>(nidx, eidx, ew, d_e, d_v);
  k_norms<<<(N_NODES + 255) / 256, 256, 0, stream>>>(d_v, d_e, ew, dv_is, de_w);

  // ---- Chebyshev recurrence ------------------------------------------------
  k_init_acc<<<NF4_BLKS, 256, 0, stream>>>((const float4*)x, theta, (float4*)acc);

  // k = 1 : T1 = -prop(x)
  hipMemsetAsync(m_e, 0, (size_t)N_EDGES * F_IN * 4, stream);
  k_spmm_v2e<<<SP_BLKS, 256, 0, stream>>>(x, nidx, eidx, dv_is, m_e);
  hipMemsetAsync(p, 0, (size_t)N_NODES * F_IN * 4, stream);
  k_spmm_e2v<<<SP_BLKS, 256, 0, stream>>>(m_e, nidx, eidx, de_w, p);
  k_combine1<<<NF4_BLKS, 256, 0, stream>>>((const float4*)p, dv_is, theta,
                                           (float4*)acc, (float4*)t0);

  // k = 2..10 : Tk = -2*prop(T_{k-1}) - T_{k-2}
  const float* Tprev = x;
  float* Tcur  = t0;
  float* Tnext = t1;
  for (int k = 2; k <= K_CHEB; ++k) {
    hipMemsetAsync(m_e, 0, (size_t)N_EDGES * F_IN * 4, stream);
    k_spmm_v2e<<<SP_BLKS, 256, 0, stream>>>(Tcur, nidx, eidx, dv_is, m_e);
    hipMemsetAsync(p, 0, (size_t)N_NODES * F_IN * 4, stream);
    k_spmm_e2v<<<SP_BLKS, 256, 0, stream>>>(m_e, nidx, eidx, de_w, p);
    k_combine<<<NF4_BLKS, 256, 0, stream>>>((const float4*)p, dv_is, theta, k,
                                            (const float4*)Tprev, (float4*)Tnext,
                                            (float4*)acc);
    float* oldCur = Tcur;
    Tprev = oldCur;   // T_{k-1}
    Tcur  = Tnext;    // T_k
    Tnext = oldCur;   // overwrite T_{k-1} buffer next iter (elementwise-safe)
  }

  // ---- dense layer: bf16x3 WMMA GEMM + bias + relu ------------------------
  k_bfsplit<<<(N_NODES * F_IN + 255) / 256, 256, 0, stream>>>(acc, a_hi, a_lo,
                                                              N_NODES * F_IN);
  k_bfsplit<<<(HID * F_IN + 255) / 256, 256, 0, stream>>>(w, w_hi, w_lo,
                                                          HID * F_IN);
  const int TILES = (N_NODES / 16) * (HID / 16);     // 100,000
  k_gemm_wmma<<<TILES / 8, 256, 0, stream>>>(a_hi, a_lo, w_hi, w_lo, bias, out);
}
